// NAGCrossAttnProcessor_23725399343122
// MI455X (gfx1250) — matrix operand order repaired and checked by reading, not compiled, not run on hardware
//
#include <hip/hip_runtime.h>
#include <hip/hip_bf16.h>

// ---------------------------------------------------------------------------
// NAG cross-attention for MI455X (gfx1250), bf16 WMMA pipeline.
//   G=2 E=2 H=24 D=128 QL=1024 KV=512 DIM=3072 CDIM=4096, ALPHA=1, EPS=1e-6
// Stages:
//   1) convert activations fp32->bf16; weights fp32->bf16 TRANSPOSED (Wt[n][k])
//   2) Q/K/V projections via v_wmma_f32_16x16x32_bf16 with DOUBLE-BUFFERED
//      LDS staging through GLOBAL_LOAD_ASYNC_TO_LDS_B128 (+ s_wait_asynccnt):
//      tile i+1 streams into LDS while tile i's WMMAs run.
//   3) fused attention: S=Q.K^T (WMMA) -> softmax -> NAG refine -> P.V (WMMA)
//   4) out = attn@Wo + bo, fp32 -> d_out
// ---------------------------------------------------------------------------

typedef __attribute__((ext_vector_type(16))) __bf16 v16bf;
typedef __attribute__((ext_vector_type(8)))  float  v8f;

namespace cfg {
constexpr int G = 2, E = 2, H = 24, D = 128;
constexpr int QL = 1024, KV = 512;
constexpr int DIM = H * D;        // 3072
constexpr int CDIM = 4096;
constexpr int B = G * E;          // 4
constexpr float SCALE = 0.08838834764831845f;   // 1/sqrt(128)
constexpr float INV1PE = 1.0f / (1.0f + 1e-6f); // 1/(1+EPS)
}

static __device__ __forceinline__ __bf16 f2bf(float f) {
  unsigned u = __builtin_bit_cast(unsigned, f);
  u += 0x7FFFu + ((u >> 16) & 1u);               // round to nearest even
  unsigned short s = (unsigned short)(u >> 16);
  return __builtin_bit_cast(__bf16, s);
}

static __device__ __forceinline__ v8f zero8() {
  v8f z = {0.f, 0.f, 0.f, 0.f, 0.f, 0.f, 0.f, 0.f};
  return z;
}

// --- CDNA5 async global->LDS copy (ASYNCcnt). Per ISA 10.2, a generic
// address hitting the LDS aperture uses addr[31:0] as the LDS byte offset,
// so the low 32 bits of a __shared__ pointer are the VDST operand.
static __device__ __forceinline__ void async_copy_b128(const void* gptr, void* lptr) {
  const unsigned ldsoff = (unsigned)(uintptr_t)lptr;
  asm volatile("global_load_async_to_lds_b128 %0, %1, off"
               :: "v"(ldsoff), "v"((unsigned long long)(uintptr_t)gptr)
               : "memory");
}
static __device__ __forceinline__ void wait_async0() {
  asm volatile("s_wait_asynccnt 0x0" ::: "memory");
}

// K index inside a 16x32 bf16 A fragment: VGPR j (j=0..7, two K each),
// lane-half h. Contiguous order (matches B): K = 16*h + 2*j.
static __device__ __forceinline__ int afrag_k(int j, int h) {
  return (h << 4) + (j << 1);
}

// A fragment (16x32, bf16) from row-major matrix, row stride ld (elements).
static __device__ __forceinline__ v16bf load_a_frag(const __bf16* p, int ld,
                                                    int row0, int k0, int lane) {
  const int m = row0 + (lane & 15);
  const int h = lane >> 4;
  const __bf16* rp = p + (size_t)m * ld + k0;
  v16bf a;
#pragma unroll
  for (int j = 0; j < 8; ++j) {
    const int k = afrag_k(j, h);
    a[2 * j]     = rp[k];
    a[2 * j + 1] = rp[k + 1];
  }
  return a;
}

// B fragment (32x16, bf16) where element (k,n) lives at p[n*ldT + k]
// (n-major / pre-transposed storage). Lane n = lane&15; VGPR j holds
// K = 16*h + 2j, 2j+1 with h = lane>>4.
static __device__ __forceinline__ v16bf load_b_frag_t(const __bf16* p, int ldT,
                                                      int k0, int n0, int lane) {
  const int n = n0 + (lane & 15);
  const int h = lane >> 4;
  const __bf16* cp = p + (size_t)n * ldT + k0 + (h << 4);
  v16bf b;
#pragma unroll
  for (int j = 0; j < 8; ++j) {
    b[2 * j]     = cp[2 * j];
    b[2 * j + 1] = cp[2 * j + 1];
  }
  return b;
}

static __device__ __forceinline__ v8f wmma_bf16(const v16bf& a, const v16bf& b,
                                                const v8f& c) {
  return __builtin_amdgcn_wmma_f32_16x16x32_bf16(
      /*neg_a=*/false, a, /*neg_b=*/false, b,
      /*c_mod=*/(short)0, c, /*reuse_a=*/false, /*reuse_b=*/false);
}

// --------------------------------------------------------------------------
// Stage 1: conversions
// --------------------------------------------------------------------------
__global__ void cvt_bf16_kernel(const float* __restrict__ in,
                                __bf16* __restrict__ out, size_t n) {
  size_t i = (size_t)blockIdx.x * blockDim.x + threadIdx.x;
  const size_t stride = (size_t)gridDim.x * blockDim.x;
  for (; i < n; i += stride) out[i] = f2bf(in[i]);
}

// out[n*K + k] = bf16(in[k*N + n]); K,N multiples of 32. block(32,8).
__global__ void transpose_bf16_kernel(const float* __restrict__ in,
                                      __bf16* __restrict__ out, int K, int N) {
  __shared__ float tile[32][33];
  const int kb = blockIdx.y * 32, nb = blockIdx.x * 32;
  const int tx = threadIdx.x, ty = threadIdx.y;
#pragma unroll
  for (int i = ty; i < 32; i += 8)
    tile[i][tx] = in[(size_t)(kb + i) * N + (nb + tx)];
  __syncthreads();
#pragma unroll
  for (int i = ty; i < 32; i += 8)
    out[(size_t)(nb + i) * K + (kb + tx)] = f2bf(tile[tx][i]);
}

// --------------------------------------------------------------------------
// Stage 2/4: WMMA GEMM, double-buffered async-LDS staging.
//   C[M,N] = A[M,K] @ W[K,N] + bias,  A row-major (ld=K), Wt[n*K+k].
//   mode 0: out bf16 row-major
//   mode 1: out bf16 scattered as vt[e][n][kv]   (rows are e*KV+kv, N=DIM)
//   mode 2: out fp32 row-major
// grid = (N/128, M/64), block = 256 (8 waves; wave = 16 rows x 64 cols).
// k-tile = 64: lA 64x64 (8KB) + lB 128x64 (16KB), x2 buffers = 48KB LDS.
// Tile i+1 is async-copied while tile i's 8 WMMAs/wave execute.
// --------------------------------------------------------------------------
__global__ void gemm_wmma_kernel(const __bf16* __restrict__ A,
                                 const __bf16* __restrict__ Wt,
                                 const float* __restrict__ bias,
                                 void* __restrict__ out,
                                 int M, int N, int K, int mode) {
  __shared__ __bf16 lA[2][64 * 64];    // lA[s][r*64 + k]
  __shared__ __bf16 lB[2][128 * 64];   // lB[s][n*64 + k]

  const int tid = threadIdx.x;
  const int lane = tid & 31;
  const int wave = tid >> 5;
  const int rowBlk = blockIdx.y * 64;
  const int nBlk = blockIdx.x * 128;
  const int rowLoc = (wave >> 1) * 16;
  const int nLoc = (wave & 1) * 64;
  (void)M;

  // stage one 64-wide k-tile into buffer s (16B chunks; 6 asyncs/thread)
  auto stage = [&](int s, int k0) {
#pragma unroll
    for (int i = 0; i < 2; ++i) {                 // A: 512 chunks
      const int c = tid + 256 * i;
      const int r = c >> 3, ko = (c & 7) * 8;
      async_copy_b128(A + (size_t)(rowBlk + r) * K + k0 + ko,
                      &lA[s][r * 64 + ko]);
    }
#pragma unroll
    for (int i = 0; i < 4; ++i) {                 // B: 1024 chunks
      const int c = tid + 256 * i;
      const int n = c >> 3, ko = (c & 7) * 8;
      async_copy_b128(Wt + (size_t)(nBlk + n) * K + k0 + ko,
                      &lB[s][n * 64 + ko]);
    }
  };

  v8f acc[4];
#pragma unroll
  for (int t = 0; t < 4; ++t) acc[t] = zero8();

  stage(0, 0);
  int s = 0;
  for (int k0 = 0; k0 < K; k0 += 64) {
    wait_async0();        // tile at k0 has landed (issued last iteration)
    __syncthreads();
    if (k0 + 64 < K) stage(s ^ 1, k0 + 64);  // overlap with WMMAs below

#pragma unroll
    for (int kk = 0; kk < 64; kk += 32) {
      const v16bf a = load_a_frag(lA[s], 64, rowLoc, kk, lane);
#pragma unroll
      for (int t = 0; t < 4; ++t) {
        const v16bf b = load_b_frag_t(lB[s], 64, kk, nLoc + 16 * t, lane);
        acc[t] = wmma_bf16(a, b, acc[t]);
      }
    }
    __syncthreads();      // all reads of buffer s done before it is re-staged
    s ^= 1;
  }

  const int rowb = rowBlk + rowLoc + ((lane >> 4) << 3);
#pragma unroll
  for (int t = 0; t < 4; ++t) {
    const int n = nBlk + nLoc + 16 * t + (lane & 15);
    const float bn = bias[n];
    if (mode == 0) {
      __bf16* o = (__bf16*)out;
#pragma unroll
      for (int i = 0; i < 8; ++i)
        o[(size_t)(rowb + i) * N + n] = f2bf(acc[t][i] + bn);
    } else if (mode == 1) {
      __bf16* o = (__bf16*)out;  // vt[e][n][kv], N == DIM
#pragma unroll
      for (int i = 0; i < 8; ++i) {
        const int rg = rowb + i;
        const int e = rg / cfg::KV, kv = rg % cfg::KV;
        o[((size_t)e * N + n) * cfg::KV + kv] = f2bf(acc[t][i] + bn);
      }
    } else {
      float* o = (float*)out;
#pragma unroll
      for (int i = 0; i < 8; ++i)
        o[(size_t)(rowb + i) * N + n] = acc[t][i] + bn;
    }
  }
}

// --------------------------------------------------------------------------
// Stage 3: fused attention + NAG.
// One wave per (g, h, 16-row q strip). LDS holds fp32 probs for both e
// (e=1 refinement needs e=0 probs). grid = (QL/16, H, G), block = 32.
// --------------------------------------------------------------------------
__global__ void attn_nag_kernel(const __bf16* __restrict__ q_bf,
                                const __bf16* __restrict__ k_bf,
                                const __bf16* __restrict__ vt,
                                __bf16* __restrict__ attn_out) {
  using namespace cfg;
  __shared__ float P[2][16][KV];  // 64 KB

  const int lane = threadIdx.x;
  const int row0 = blockIdx.x * 16;
  const int h = blockIdx.y;
  const int g = blockIdx.z;

  const int r = lane & 15;       // row owned for reductions
  const int half = lane >> 4;    // which 256-col half this lane scans
  const int dcol = lane & 15;    // D-tile column
  const int drow = (lane >> 4) << 3;

  // ---- scores + softmax, per e ----
  for (int e = 0; e < 2; ++e) {
    const __bf16* Qp = q_bf + ((size_t)(g * E + e) * QL) * DIM + h * D;
    const __bf16* Kp = k_bf + ((size_t)e * KV) * DIM + h * D;  // (k,n)=(d,kv) at Kp[n*DIM+k]

    for (int nt = 0; nt < KV / 16; ++nt) {
      v8f c = zero8();
#pragma unroll
      for (int kk = 0; kk < 4; ++kk) {
        const v16bf a = load_a_frag(Qp, DIM, row0, kk * 32, lane);
        const v16bf b = load_b_frag_t(Kp, DIM, kk * 32, nt * 16, lane);
        c = wmma_bf16(a, b, c);
      }
      const int col = nt * 16 + dcol;
#pragma unroll
      for (int i = 0; i < 8; ++i) P[e][drow + i][col] = c[i] * SCALE;
    }
    __syncthreads();  // single wave: S_NOP, keeps DS ordering obvious

    // row softmax: lane r handles half of row r, merge halves via shfl_xor 16
    float mx = -3.4e38f;
    for (int i = 0; i < KV / 2; ++i)
      mx = fmaxf(mx, P[e][r][half * (KV / 2) + i]);
    mx = fmaxf(mx, __shfl_xor(mx, 16));
    float sum = 0.f;
    for (int i = 0; i < KV / 2; ++i) {
      const int c2 = half * (KV / 2) + i;
      const float v = __expf(P[e][r][c2] - mx);
      P[e][r][c2] = v;
      sum += v;
    }
    sum += __shfl_xor(sum, 16);
    const float inv = 1.0f / sum;
    for (int i = 0; i < KV / 2; ++i) P[e][r][half * (KV / 2) + i] *= inv;
    __syncthreads();
  }

  // ---- NAG refine e=1:  w1 = softmax(2*n1 - n0), n = p/(1+EPS) ----
  {
    float mx = -3.4e38f;
    for (int i = 0; i < KV / 2; ++i) {
      const int c2 = half * (KV / 2) + i;
      const float z = (2.0f * P[1][r][c2] - P[0][r][c2]) * INV1PE;
      P[1][r][c2] = z;
      mx = fmaxf(mx, z);
    }
    mx = fmaxf(mx, __shfl_xor(mx, 16));
    float sum = 0.f;
    for (int i = 0; i < KV / 2; ++i) {
      const int c2 = half * (KV / 2) + i;
      const float v = __expf(P[1][r][c2] - mx);
      P[1][r][c2] = v;
      sum += v;
    }
    sum += __shfl_xor(sum, 16);
    const float inv = 1.0f / sum;
    for (int i = 0; i < KV / 2; ++i) P[1][r][half * (KV / 2) + i] *= inv;
    __syncthreads();
  }

  // ---- O = P @ V ----
  const int am = lane & 15;
  const int ah = lane >> 4;
  for (int e = 0; e < 2; ++e) {
    const __bf16* Vp = vt + ((size_t)e * DIM + h * D) * KV;  // (k,n)=(kv,d) at Vp[n*KV+k]
    __bf16* Op = attn_out + ((size_t)(g * E + e) * QL + row0) * DIM + h * D;
#pragma unroll
    for (int t = 0; t < D / 16; ++t) {
      v8f c = zero8();
      for (int kk = 0; kk < KV / 32; ++kk) {
        v16bf a;
#pragma unroll
        for (int j = 0; j < 8; ++j) {
          const int k = kk * 32 + afrag_k(j, ah);
          a[2 * j]     = f2bf(P[e][am][k]);
          a[2 * j + 1] = f2bf(P[e][am][k + 1]);
        }
        const v16bf b = load_b_frag_t(Vp, KV, kk * 32, t * 16, lane);
        c = wmma_bf16(a, b, c);
      }
      const int n = t * 16 + dcol;
#pragma unroll
      for (int i = 0; i < 8; ++i)
        Op[(size_t)(drow + i) * DIM + n] = f2bf(c[i]);
    }
  }
}

// --------------------------------------------------------------------------
// Host launcher
// --------------------------------------------------------------------------
extern "C" void kernel_launch(void* const* d_in, const int* in_sizes, int n_in,
                              void* d_out, int out_size, void* d_ws, size_t ws_size,
                              hipStream_t stream) {
  using namespace cfg;
  (void)in_sizes; (void)n_in; (void)out_size; (void)ws_size;

  const float* hs  = (const float*)d_in[0];
  const float* ehs = (const float*)d_in[1];
  const float* Wq  = (const float*)d_in[2];
  const float* bq  = (const float*)d_in[3];
  const float* Wk  = (const float*)d_in[4];
  const float* bk  = (const float*)d_in[5];
  const float* Wv  = (const float*)d_in[6];
  const float* bv  = (const float*)d_in[7];
  const float* Wo  = (const float*)d_in[8];
  const float* bo  = (const float*)d_in[9];

  // workspace carve-out (~185 MB total)
  size_t off = 0;
  auto take = [&](size_t bytes) -> void* {
    void* p = (char*)d_ws + off;
    off += (bytes + 255) & ~(size_t)255;
    return p;
  };
  __bf16* hs_bf  = (__bf16*)take((size_t)B * QL * DIM * 2);
  __bf16* ehs_bf = (__bf16*)take((size_t)E * KV * CDIM * 2);
  __bf16* Wq_t   = (__bf16*)take((size_t)DIM * DIM * 2);
  __bf16* Wk_t   = (__bf16*)take((size_t)DIM * CDIM * 2);
  __bf16* Wv_t   = (__bf16*)take((size_t)DIM * CDIM * 2);
  __bf16* Wo_t   = (__bf16*)take((size_t)DIM * DIM * 2);
  __bf16* q_bf   = (__bf16*)take((size_t)B * QL * DIM * 2);
  __bf16* k_bf   = (__bf16*)take((size_t)E * KV * DIM * 2);
  __bf16* vt_bf  = (__bf16*)take((size_t)E * DIM * KV * 2);
  __bf16* at_bf  = (__bf16*)take((size_t)B * QL * DIM * 2);

  // stage 1: conversions
  cvt_bf16_kernel<<<2048, 256, 0, stream>>>(hs, hs_bf, (size_t)B * QL * DIM);
  cvt_bf16_kernel<<<2048, 256, 0, stream>>>(ehs, ehs_bf, (size_t)E * KV * CDIM);
  {
    dim3 blk(32, 8);
    transpose_bf16_kernel<<<dim3(DIM / 32, DIM / 32), blk, 0, stream>>>(Wq, Wq_t, DIM, DIM);
    transpose_bf16_kernel<<<dim3(DIM / 32, CDIM / 32), blk, 0, stream>>>(Wk, Wk_t, CDIM, DIM);
    transpose_bf16_kernel<<<dim3(DIM / 32, CDIM / 32), blk, 0, stream>>>(Wv, Wv_t, CDIM, DIM);
    transpose_bf16_kernel<<<dim3(DIM / 32, DIM / 32), blk, 0, stream>>>(Wo, Wo_t, DIM, DIM);
  }

  // stage 2: projections
  gemm_wmma_kernel<<<dim3(DIM / 128, (B * QL) / 64), 256, 0, stream>>>(
      hs_bf, Wq_t, bq, q_bf, B * QL, DIM, DIM, 0);
  gemm_wmma_kernel<<<dim3(DIM / 128, (E * KV) / 64), 256, 0, stream>>>(
      ehs_bf, Wk_t, bk, k_bf, E * KV, DIM, CDIM, 0);
  gemm_wmma_kernel<<<dim3(DIM / 128, (E * KV) / 64), 256, 0, stream>>>(
      ehs_bf, Wv_t, bv, vt_bf, E * KV, DIM, CDIM, 1);

  // stage 3: fused attention + NAG
  attn_nag_kernel<<<dim3(QL / 16, H, G), 32, 0, stream>>>(q_bf, k_bf, vt_bf, at_bf);

  // stage 4: output projection -> fp32 d_out
  gemm_wmma_kernel<<<dim3(DIM / 128, (B * QL) / 64), 256, 0, stream>>>(
      at_bf, Wo_t, bo, d_out, B * QL, DIM, DIM, 2);
}
